// MultiHeadAttention_17480516895333
// MI455X (gfx1250) — matrix operand order repaired and checked by reading, not compiled, not run on hardware
//
#include <hip/hip_runtime.h>
#include <hip/hip_bf16.h>

typedef _Float16 f16;
typedef __attribute__((ext_vector_type(16))) _Float16 v16h;
typedef __attribute__((ext_vector_type(8)))  _Float16 v8h;
typedef __attribute__((ext_vector_type(4)))  _Float16 v4h;
typedef __attribute__((ext_vector_type(8)))  float    v8f;

union Frag { v16h v; v8h h[2]; };
union PackH { _Float16 h[2]; unsigned u; };

#define LOG2E 1.4426950408889634f

static __device__ __forceinline__ v8f wmma_f16(v16h a, v16h b, v8f c) {
    // D = A(16x32 f16) * B(32x16 f16) + C(16x16 f32)
    return __builtin_amdgcn_wmma_f32_16x16x32_f16(false, a, false, b, (short)0, c, false, false);
}

// ---------------------------------------------------------------------------
// Kernel 1: QKV = x @ W_qkv + b_qkv   ([4096,256] x [256,768])
// Q (scaled by log2e/sqrt(32)) and K go to augmented rows [B,H,N,96] (first
// 32 dims); V goes transposed [B,H,32,N] so attention V-frags are contiguous.
// ---------------------------------------------------------------------------
__global__ __launch_bounds__(128) void qkv_gemm_kernel(
    const float* __restrict__ x, const float* __restrict__ Wqkv,
    const float* __restrict__ bqkv,
    f16* __restrict__ qa, f16* __restrict__ ka, f16* __restrict__ vtws)
{
    __shared__ __align__(16) _Float16 lsA[64][32];   // [m][k]
    __shared__ __align__(16) _Float16 lsBt[64][32];  // [n][k]
    const int m0 = blockIdx.x * 64;
    const int n0 = blockIdx.y * 64;
    const int t = threadIdx.x;
    const int wave = t >> 5, lane = t & 31;
    const int lm = lane & 15, hi = lane >> 4;
    const int k0 = hi * 8;    // unused (A-frag uses kb too)
    const int kb = hi * 16;   // B-frag k chunk base

    v8f acc[4] = {};
    for (int kk = 0; kk < 256; kk += 32) {
        __syncthreads();
        const float4* xv = (const float4*)x;          // pitch 64 float4
#pragma unroll
        for (int i = 0; i < 4; ++i) {                 // A tile 64x32: b128 loads
            int idx = t + i * 128;                    // 0..511 float4s
            int m = idx >> 3, q = idx & 7;
            float4 f = xv[(m0 + m) * 64 + (kk >> 2) + q];
            v4h hv;
            hv[0] = (_Float16)f.x; hv[1] = (_Float16)f.y;
            hv[2] = (_Float16)f.z; hv[3] = (_Float16)f.w;
            *(v4h*)&lsA[m][q * 4] = hv;               // ds_store_b64
        }
#pragma unroll
        for (int i = 0; i < 4; ++i) {                 // B tile 32x64, transposed
            int idx = t + i * 128;                    // 0..511 float4s
            int kr = idx >> 4, nq = idx & 15;
            float4 f = *(const float4*)&Wqkv[(kk + kr) * 768 + n0 + nq * 4];
            int n = nq * 4;
            lsBt[n][kr]     = (_Float16)f.x;
            lsBt[n + 1][kr] = (_Float16)f.y;
            lsBt[n + 2][kr] = (_Float16)f.z;
            lsBt[n + 3][kr] = (_Float16)f.w;
        }
        __syncthreads();
        Frag a;
        a.h[0] = *(const v8h*)&lsA[16 * wave + lm][kb];
        a.h[1] = *(const v8h*)&lsA[16 * wave + lm][kb + 8];
#pragma unroll
        for (int c = 0; c < 4; ++c) {
            Frag bf;
            bf.h[0] = *(const v8h*)&lsBt[c * 16 + lm][kb];
            bf.h[1] = *(const v8h*)&lsBt[c * 16 + lm][kb + 8];
            acc[c] = wmma_f16(a.v, bf.v, acc[c]);
        }
    }

    // epilogue: which-matrix is provably uniform (n0 is 64-aligned, intra-tile
    // offsets < 64 cannot cross a 256 boundary) -> one scalar branch
    const float qscale = 0.17677669529663687f * LOG2E;  // log2e/sqrt(32)
    const int which = n0 >> 8;
    const int cbase = n0 & 255;
    if (which == 0) {
#pragma unroll
        for (int c = 0; c < 4; ++c) {
            int cc = cbase + c * 16 + lm;
            int h = cc >> 5, d = cc & 31;
            float bias = bqkv[n0 + c * 16 + lm];
#pragma unroll
            for (int r = 0; r < 8; ++r) {
                int row = m0 + 16 * wave + r + 8 * hi;
                int b = row >> 11, nn = row & 2047;
                qa[((b * 8 + h) * 2048 + nn) * 96 + d] =
                    (_Float16)((acc[c][r] + bias) * qscale);
            }
        }
    } else if (which == 1) {
#pragma unroll
        for (int c = 0; c < 4; ++c) {
            int cc = cbase + c * 16 + lm;
            int h = cc >> 5, d = cc & 31;
            float bias = bqkv[n0 + c * 16 + lm];
#pragma unroll
            for (int r = 0; r < 8; ++r) {
                int row = m0 + 16 * wave + r + 8 * hi;
                int b = row >> 11, nn = row & 2047;
                ka[((b * 8 + h) * 2048 + nn) * 96 + d] = (_Float16)(acc[c][r] + bias);
            }
        }
    } else {
#pragma unroll
        for (int c = 0; c < 4; ++c) {
            int cc = cbase + c * 16 + lm;
            int h = cc >> 5, d = cc & 31;
            float bias = bqkv[n0 + c * 16 + lm];
#pragma unroll
            for (int r = 0; r < 8; ++r) {
                int row = m0 + 16 * wave + r + 8 * hi;
                int b = row >> 11, nn = row & 2047;
                vtws[(b * 8 + h) * 65536 + d * 2048 + nn] = (_Float16)(acc[c][r] + bias);
            }
        }
    }
}

// ---------------------------------------------------------------------------
// Kernel 2: fill augmented dims 32..79.
//   Q aug: 3x 16-wide one-hot of position components (value log2e, so the
//          whole score is pre-multiplied by log2e for exp2-based softmax)
//   K aug: shifted per-head table slices  w[a] = table[a - p + 32]
// so q_aug . k_aug = log2e * (td[pi-pj+32] + th[..] + tw[..])  (exact rel
// bias; |rel| <= 15 so the reference clamp never fires).
// ---------------------------------------------------------------------------
__global__ __launch_bounds__(256) void aug_fill_kernel(
    const int* __restrict__ pos32, const float* __restrict__ td,
    const float* __restrict__ th, const float* __restrict__ tw,
    f16* __restrict__ qa, f16* __restrict__ ka)
{
    int idx = blockIdx.x * 256 + threadIdx.x;     // over B*H*N = 32768 rows
    int nn = idx & 2047;
    int bh = idx >> 11;
    int h = bh & 7, b = bh >> 3;
    int pi = (b * 2048 + nn) * 3;
    int p0 = pos32[(pi + 0) * 2];                 // int64 inputs: low words
    int p1 = pos32[(pi + 1) * 2];
    int p2 = pos32[(pi + 2) * 2];
    f16* qrow = qa + idx * 96 + 32;
    f16* krow = ka + idx * 96 + 32;
    const _Float16 one = (_Float16)LOG2E;
    const _Float16 zero = (_Float16)0.0f;
#pragma unroll
    for (int a = 0; a < 16; ++a) {
        qrow[a]      = (a == p0) ? one : zero;
        qrow[16 + a] = (a == p1) ? one : zero;
        qrow[32 + a] = (a == p2) ? one : zero;
        krow[a]      = (_Float16)td[(a - p0 + 32) * 8 + h];
        krow[16 + a] = (_Float16)th[(a - p1 + 32) * 8 + h];
        krow[32 + a] = (_Float16)tw[(a - p2 + 32) * 8 + h];
    }
}

// ---------------------------------------------------------------------------
// Kernel 3: flash attention, bias folded into the GEMM (k-dim 96 = 3x32).
// One wave per (b,h, 16-row Q tile); 64 key tiles of 32 tokens.
// S0 columns = even tokens, S1 columns = odd tokens: each lane's two P values
// are adjacent tokens -> one packed b32 LDS store, and the P/V fragments keep
// natural contiguous k-order. Scores are bounded, softmax is shift-invariant
// => no running max; p = exp2(s) (log2e folded into Q).
// ---------------------------------------------------------------------------
__global__ __launch_bounds__(32) void attn_kernel(
    const f16* __restrict__ qa, const f16* __restrict__ ka,
    const f16* __restrict__ vtws, f16* __restrict__ attn_out)
{
    __shared__ __align__(16) _Float16 lsP[16 * 32];  // P tile relayout buffer
    const int lane = threadIdx.x;
    const int lm = lane & 15, hi = lane >> 4;
    const int k0 = hi * 8, kb = hi * 16;
    const int blk = blockIdx.x;
    const int tile = blk & 127;        // N/16
    const int bh = blk >> 7;
    const int b = bh >> 3, h = bh & 7;

    // Q A-fragments for the 3 k-chunks (row m = lm): contiguous 16B loads
    const f16* qbase = qa + (bh * 2048 + tile * 16) * 96;
    Frag aq[3];
#pragma unroll
    for (int c = 0; c < 3; ++c) {
        aq[c].h[0] = *(const v8h*)(qbase + lm * 96 + 32 * c + kb);
        aq[c].h[1] = *(const v8h*)(qbase + lm * 96 + 32 * c + kb + 8);
    }

    v8f o0 = {}, o1 = {};
    float lrun[8];
#pragma unroll
    for (int r = 0; r < 8; ++r) lrun[r] = 0.0f;

    const f16* kbase = ka + bh * 2048 * 96;   // [N,96]
    const f16* vbase = vtws + bh * 65536;     // [32,N]
    unsigned* lsPw = (unsigned*)lsP;

    for (int kt = 0; kt < 64; ++kt) {
        const int tok0 = kt * 32;
        // S0 cols = even tokens, S1 cols = odd tokens (per-lane K rows)
        const f16* kr0 = kbase + (tok0 + 2 * lm) * 96;
        const f16* kr1 = kbase + (tok0 + 2 * lm + 1) * 96;
        v8f s0, s1;
        {
            v8f z = {};
            Frag kf;
            kf.h[0] = *(const v8h*)(kr0 + kb);
            kf.h[1] = *(const v8h*)(kr0 + kb + 8);
            s0 = wmma_f16(aq[0].v, kf.v, z);
            kf.h[0] = *(const v8h*)(kr0 + 32 + kb);
            kf.h[1] = *(const v8h*)(kr0 + 32 + kb + 8);
            s0 = wmma_f16(aq[1].v, kf.v, s0);
            kf.h[0] = *(const v8h*)(kr0 + 64 + kb);
            kf.h[1] = *(const v8h*)(kr0 + 64 + kb + 8);
            s0 = wmma_f16(aq[2].v, kf.v, s0);
            kf.h[0] = *(const v8h*)(kr1 + kb);
            kf.h[1] = *(const v8h*)(kr1 + kb + 8);
            s1 = wmma_f16(aq[0].v, kf.v, z);
            kf.h[0] = *(const v8h*)(kr1 + 32 + kb);
            kf.h[1] = *(const v8h*)(kr1 + 32 + kb + 8);
            s1 = wmma_f16(aq[1].v, kf.v, s1);
            kf.h[0] = *(const v8h*)(kr1 + 64 + kb);
            kf.h[1] = *(const v8h*)(kr1 + 64 + kb + 8);
            s1 = wmma_f16(aq[2].v, kf.v, s1);
        }

        // p = exp2(s); per-lane partial row sums; packed b32 relayout stores
#pragma unroll
        for (int r = 0; r < 8; ++r) {
            PackH pk;
            float p0 = exp2f(s0[r]);
            float p1 = exp2f(s1[r]);
            lrun[r] += p0 + p1;
            pk.h[0] = (_Float16)p0;                  // token 2*lm
            pk.h[1] = (_Float16)p1;                  // token 2*lm+1
            lsPw[(r + 8 * hi) * 16 + lm] = pk.u;     // ds_store_b32
        }
        __syncthreads();
        Frag ap;
        ap.h[0] = *(const v8h*)&lsP[lm * 32 + kb];
        ap.h[1] = *(const v8h*)&lsP[lm * 32 + kb + 8];

        // V B-fragments: contiguous thanks to transposed V layout [d][token]
        Frag vf0, vf1;
        vf0.h[0] = *(const v8h*)(vbase + lm * 2048 + tok0 + kb);
        vf0.h[1] = *(const v8h*)(vbase + lm * 2048 + tok0 + kb + 8);
        vf1.h[0] = *(const v8h*)(vbase + (16 + lm) * 2048 + tok0 + kb);
        vf1.h[1] = *(const v8h*)(vbase + (16 + lm) * 2048 + tok0 + kb + 8);
        o0 = wmma_f16(ap.v, vf0.v, o0);
        o1 = wmma_f16(ap.v, vf1.v, o1);
        __syncthreads();
    }

    // single final row-sum reduction across the 16-lane half, then normalize
#pragma unroll
    for (int r = 0; r < 8; ++r) {
        float l = lrun[r];
        l += __shfl_xor(l, 1, 32);
        l += __shfl_xor(l, 2, 32);
        l += __shfl_xor(l, 4, 32);
        l += __shfl_xor(l, 8, 32);
        float inv = 1.0f / l;
        int tok = tile * 16 + r + 8 * hi;
        int base = (b * 2048 + tok) * 256 + h * 32;
        attn_out[base + lm]      = (_Float16)(o0[r] * inv);
        attn_out[base + 16 + lm] = (_Float16)(o1[r] * inv);
    }
}

// ---------------------------------------------------------------------------
// Kernel 4: out = attn @ W_proj + b_proj  ([4096,256] x [256,256]) -> f32
// ---------------------------------------------------------------------------
__global__ __launch_bounds__(128) void proj_gemm_kernel(
    const f16* __restrict__ attn, const float* __restrict__ Wp,
    const float* __restrict__ bp, float* __restrict__ out)
{
    __shared__ __align__(16) _Float16 lsA[64][32];
    __shared__ __align__(16) _Float16 lsBt[64][32];
    const int m0 = blockIdx.x * 64;
    const int n0 = blockIdx.y * 64;
    const int t = threadIdx.x;
    const int wave = t >> 5, lane = t & 31;
    const int lm = lane & 15, hi = lane >> 4;
    const int k0 = hi * 8, kb = hi * 16;

    v8f acc[4] = {};
    uint4* lsA4 = (uint4*)&lsA[0][0];
    const uint4* aw4 = (const uint4*)attn;            // pitch 32 uint4 per row
    for (int kk = 0; kk < 256; kk += 32) {
        __syncthreads();
        {                                             // A tile: 1 b128 per thread
            int m = t >> 1, w = t & 1;                // 64 rows x 2 uint4
            lsA4[m * 2 + w] = aw4[(m0 + m) * 32 + (kk >> 3) + w];
        }
#pragma unroll
        for (int i = 0; i < 4; ++i) {                 // B tile transposed
            int idx = t + i * 128;
            int kr = idx >> 4, nq = idx & 15;
            float4 f = *(const float4*)&Wp[(kk + kr) * 256 + n0 + nq * 4];
            int n = nq * 4;
            lsBt[n][kr]     = (_Float16)f.x;
            lsBt[n + 1][kr] = (_Float16)f.y;
            lsBt[n + 2][kr] = (_Float16)f.z;
            lsBt[n + 3][kr] = (_Float16)f.w;
        }
        __syncthreads();
        Frag a;
        a.h[0] = *(const v8h*)&lsA[16 * wave + lm][kb];
        a.h[1] = *(const v8h*)&lsA[16 * wave + lm][kb + 8];
#pragma unroll
        for (int c = 0; c < 4; ++c) {
            Frag bf;
            bf.h[0] = *(const v8h*)&lsBt[c * 16 + lm][kb];
            bf.h[1] = *(const v8h*)&lsBt[c * 16 + lm][kb + 8];
            acc[c] = wmma_f16(a.v, bf.v, acc[c]);
        }
    }
#pragma unroll
    for (int c = 0; c < 4; ++c) {
        int col = n0 + c * 16 + lm;
        float bias = bp[col];
#pragma unroll
        for (int r = 0; r < 8; ++r) {
            int row = m0 + 16 * wave + r + 8 * hi;
            out[row * 256 + col] = acc[c][r] + bias;
        }
    }
}

// ---------------------------------------------------------------------------
extern "C" void kernel_launch(void* const* d_in, const int* in_sizes, int n_in,
                              void* d_out, int out_size, void* d_ws, size_t ws_size,
                              hipStream_t stream) {
    const float* x     = (const float*)d_in[0];
    const int*   pos32 = (const int*)d_in[1];   // int64 positions, read low words
    const float* Wqkv  = (const float*)d_in[2];
    const float* bqkv  = (const float*)d_in[3];
    const float* Wproj = (const float*)d_in[4];
    const float* bproj = (const float*)d_in[5];
    const float* td    = (const float*)d_in[6];
    const float* th    = (const float*)d_in[7];
    const float* tw    = (const float*)d_in[8];
    float* out = (float*)d_out;

    // workspace: Qaug | Kaug (16*2048*96 f16 each) | V^T | attn_out = 16 MB
    f16* qa = (f16*)d_ws;
    f16* ka = qa + 16 * 2048 * 96;
    f16* vt = ka + 16 * 2048 * 96;
    f16* ao = vt + (1 << 20);

    aug_fill_kernel<<<128, 256, 0, stream>>>(pos32, td, th, tw, qa, ka);
    qkv_gemm_kernel<<<dim3(64, 12), 128, 0, stream>>>(x, Wqkv, bqkv, qa, ka, vt);
    attn_kernel<<<2048, 32, 0, stream>>>(qa, ka, vt, ao);
    proj_gemm_kernel<<<dim3(64, 4), 128, 0, stream>>>(ao, Wproj, bproj, out);
}
